// L1OutUB_14422500180350
// MI455X (gfx1250) — matrix operand order repaired and checked by reading, not compiled, not run on hardware
//
#include <hip/hip_runtime.h>
#include <hip/hip_bf16.h>

typedef __attribute__((ext_vector_type(16))) __bf16 v16bf;
typedef __attribute__((ext_vector_type(8)))  __bf16 v8bf;
typedef __attribute__((ext_vector_type(8)))  float  v8f;

// ---------- small helpers ----------

// cheap tanh with correct +-1 saturation: 1 - 2/(exp(2x)+1)
__device__ __forceinline__ float fast_tanh(float x) {
  float t = __expf(2.0f * x);
  return 1.0f - 2.0f / (t + 1.0f);
}

// flat f32 -> bf16 convert
__global__ void cvt_bf16(const float* __restrict__ src,
                         __bf16* __restrict__ dst, int n) {
  int idx = blockIdx.x * blockDim.x + threadIdx.x;
  if (idx < n) dst[idx] = (__bf16)src[idx];
}

// W[K,N] f32 -> Wt[N,K] bf16  (so GEMM B-operand loads are contiguous in K)
__global__ void transpose_cvt_bf16(const float* __restrict__ W,
                                   __bf16* __restrict__ Wt, int K, int N) {
  int idx = blockIdx.x * blockDim.x + threadIdx.x; // over N*K, out-linear
  if (idx >= K * N) return;
  int n = idx / K;
  int k = idx - n * K;
  Wt[idx] = (__bf16)W[(size_t)k * N + n];
}

// ---------- WMMA GEMM: out[M,N] = act(A[M,K] @ Bt[N,K]^T + bias[N]) ----------
// A row-major bf16, Bt row-major bf16 (i.e. B[k,n] = Bt[n*K+k]).
// ACT: 0 = relu, 1 = identity, 2 = tanh.  OUT_BF16 selects output type.
// One wave computes one 16x16 tile with v_wmma_f32_16x16x32_bf16.
template <int ACT, bool OUT_BF16>
__global__ void gemm_bf16(const __bf16* __restrict__ A,
                          const __bf16* __restrict__ Bt,
                          const float* __restrict__ bias,
                          float* __restrict__ outf,
                          __bf16* __restrict__ outb,
                          int M, int N, int K) {
  const int wave = threadIdx.x >> 5;
  const int lane = threadIdx.x & 31;
  const int tiles_n = N >> 4;
  const int tile = blockIdx.x * (blockDim.x >> 5) + wave;
  if (tile >= (M >> 4) * tiles_n) return;
  const int tm = tile / tiles_n;
  const int tn = tile - tm * tiles_n;
  const int m = lane & 15;  // A-row / B-col within tile for this lane
  const int h = lane >> 4;  // lane half

  v8f acc = {};
  const __bf16* arow = A + (size_t)(tm * 16 + m) * K;
  const __bf16* brow = Bt + (size_t)(tn * 16 + m) * K;

  for (int k0 = 0; k0 < K; k0 += 32) {
    __builtin_prefetch(arow + k0 + 128, 0, 0);
    __builtin_prefetch(brow + k0 + 128, 0, 0);
    // A 16x32: lane half h holds K = k0+h*8+{0..7} (slots 0..7) and
    // K = k0+16+h*8+{0..7} (slots 8..15) -> two 16B vector loads.
    v8bf alo = *(const v8bf*)(arow + k0 + h * 8);
    v8bf ahi = *(const v8bf*)(arow + k0 + 16 + h * 8);
    v16bf a = __builtin_shufflevector(alo, ahi, 0, 1, 2, 3, 4, 5, 6, 7,
                                      8, 9, 10, 11, 12, 13, 14, 15);
    // B 32x16: same K order as A -> two 16B vector loads.
    v8bf blo = *(const v8bf*)(brow + k0 + h * 8);
    v8bf bhi = *(const v8bf*)(brow + k0 + 16 + h * 8);
    v16bf b = __builtin_shufflevector(blo, bhi, 0, 1, 2, 3, 4, 5, 6, 7,
                                      8, 9, 10, 11, 12, 13, 14, 15);
    acc = __builtin_amdgcn_wmma_f32_16x16x32_bf16(false, a, false, b,
                                                  (short)0, acc, false, false);
  }

  const int col = tn * 16 + m;
  const float bv = bias[col];
#pragma unroll
  for (int g = 0; g < 8; ++g) {
    // C/D layout: VGPR g, lane half h -> local row = h*8 + g, col = lane&15
    const int row = tm * 16 + h * 8 + g;
    float v = acc[g] + bv;
    if (ACT == 0)      v = fmaxf(v, 0.0f);
    else if (ACT == 2) v = fast_tanh(v);
    if (OUT_BF16) outb[(size_t)row * N + col] = (__bf16)v;
    else          outf[(size_t)row * N + col] = v;
  }
}

// ---------- contrastive panels ----------
//   Abuf[i, 0:128]   = mu[i]*exp(-lv[i])       Abuf[i,128:256] = -0.5*exp(-lv[i])
//   Bbuf[j, 0:128]   = y[j]                    Bbuf[j,128:256] = y[j]^2
//   c[i] = -0.5 * sum_k (mu^2 * exp(-lv) + lv)
__global__ void prep_contrast(const float* __restrict__ mu,
                              const float* __restrict__ lv,
                              const float* __restrict__ y,
                              __bf16* __restrict__ Abuf,
                              __bf16* __restrict__ Bbuf,
                              float* __restrict__ c) {
  __shared__ float red[128];
  const int i = blockIdx.x;
  const int t = threadIdx.x; // 128 threads, t == k
  const float mv  = mu[i * 128 + t];
  const float lvv = lv[i * 128 + t];
  const float e   = __expf(-lvv);
  Abuf[i * 256 + t]       = (__bf16)(mv * e);
  Abuf[i * 256 + 128 + t] = (__bf16)(-0.5f * e);
  const float yv = y[i * 128 + t];
  Bbuf[i * 256 + t]       = (__bf16)yv;
  Bbuf[i * 256 + 128 + t] = (__bf16)(yv * yv);
  red[t] = mv * mv * e + lvv;
  __syncthreads();
  for (int s = 64; s > 0; s >>= 1) {
    if (t < s) red[t] += red[t + s];
    __syncthreads();
  }
  if (t == 0) c[i] = -0.5f * red[0];
}

// ---------- all_probs: single 512x512 (K=256) WMMA GEMM, transposed store ----
//   apT[j*512 + i] = sum_k Abuf[i,k]*Bbuf[j,k] + c[i]
__global__ void gemm_ap(const __bf16* __restrict__ Abuf, // 512 x 256
                        const __bf16* __restrict__ Bbuf, // 512 x 256
                        const float* __restrict__ c,
                        float* __restrict__ apT) {
  const int wave = threadIdx.x >> 5;
  const int lane = threadIdx.x & 31;
  const int tile = blockIdx.x * (blockDim.x >> 5) + wave;
  const int tiles_n = 32; // 512/16
  if (tile >= 32 * tiles_n) return;
  const int tm = tile / tiles_n;
  const int tn = tile - tm * tiles_n;
  const int m = lane & 15;
  const int h = lane >> 4;
  const int K = 256;

  v8f acc = {};
  const __bf16* arow = Abuf + (size_t)(tm * 16 + m) * K;
  const __bf16* brow = Bbuf + (size_t)(tn * 16 + m) * K;

#pragma unroll
  for (int k0 = 0; k0 < 256; k0 += 32) {
    v8bf alo = *(const v8bf*)(arow + k0 + h * 8);
    v8bf ahi = *(const v8bf*)(arow + k0 + 16 + h * 8);
    v16bf a = __builtin_shufflevector(alo, ahi, 0, 1, 2, 3, 4, 5, 6, 7,
                                      8, 9, 10, 11, 12, 13, 14, 15);
    v8bf blo = *(const v8bf*)(brow + k0 + h * 8);
    v8bf bhi = *(const v8bf*)(brow + k0 + 16 + h * 8);
    v16bf b = __builtin_shufflevector(blo, bhi, 0, 1, 2, 3, 4, 5, 6, 7,
                                      8, 9, 10, 11, 12, 13, 14, 15);
    acc = __builtin_amdgcn_wmma_f32_16x16x32_bf16(false, a, false, b,
                                                  (short)0, acc, false, false);
  }

  const int col = tn * 16 + m; // j
#pragma unroll
  for (int g = 0; g < 8; ++g) {
    const int row = tm * 16 + h * 8 + g; // i
    apT[(size_t)col * 512 + row] = acc[g] + c[row];
  }
}

// ---------- per-sample logsumexp + final mean ----------
__global__ void col_lse(const float* __restrict__ apT,
                        float* __restrict__ contrib) {
  __shared__ float red[256];
  const int j = blockIdx.x;
  const int t = threadIdx.x;
  const float* rowp = apT + (size_t)j * 512;

  float mx = -3.4e38f;
  for (int i = t; i < 512; i += 256) {
    float v = rowp[i] + ((i == j) ? -20.0f : 0.0f);
    mx = fmaxf(mx, v);
  }
  red[t] = mx; __syncthreads();
  for (int s = 128; s > 0; s >>= 1) {
    if (t < s) red[t] = fmaxf(red[t], red[t + s]);
    __syncthreads();
  }
  const float mall = red[0];
  __syncthreads();

  float sm = 0.0f;
  for (int i = t; i < 512; i += 256) {
    float v = rowp[i] + ((i == j) ? -20.0f : 0.0f);
    sm += __expf(v - mall);
  }
  red[t] = sm; __syncthreads();
  for (int s = 128; s > 0; s >>= 1) {
    if (t < s) red[t] += red[t + s];
    __syncthreads();
  }
  if (t == 0) {
    const float negative = mall + __logf(red[0]) - __logf(511.0f);
    contrib[j] = rowp[j] - negative; // positive = ap[j,j] (unmasked)
  }
}

__global__ void final_mean(const float* __restrict__ contrib,
                           float* __restrict__ out) {
  __shared__ float red[256];
  const int t = threadIdx.x;
  red[t] = contrib[t] + contrib[t + 256];
  __syncthreads();
  for (int s = 128; s > 0; s >>= 1) {
    if (t < s) red[t] += red[t + s];
    __syncthreads();
  }
  if (t == 0) out[0] = red[0] * (1.0f / 512.0f);
}

// ---------- host side ----------
extern "C" void kernel_launch(void* const* d_in, const int* in_sizes, int n_in,
                              void* d_out, int out_size, void* d_ws, size_t ws_size,
                              hipStream_t stream) {
  const float* x   = (const float*)d_in[0]; // 512x768
  const float* y   = (const float*)d_in[1]; // 512x128
  const float* w1m = (const float*)d_in[2]; // 768x512
  const float* b1m = (const float*)d_in[3]; // 512
  const float* w2m = (const float*)d_in[4]; // 512x128
  const float* b2m = (const float*)d_in[5]; // 128
  const float* w1l = (const float*)d_in[6];
  const float* b1l = (const float*)d_in[7];
  const float* w2l = (const float*)d_in[8];
  const float* b2l = (const float*)d_in[9];

  char* ws = (char*)d_ws;
  __bf16* Xb      = (__bf16*)ws; ws += (size_t)512 * 768 * 2;
  __bf16* W1mt    = (__bf16*)ws; ws += (size_t)512 * 768 * 2; // [N=512,K=768]
  __bf16* W1lt    = (__bf16*)ws; ws += (size_t)512 * 768 * 2;
  __bf16* W2mt    = (__bf16*)ws; ws += (size_t)128 * 512 * 2; // [N=128,K=512]
  __bf16* W2lt    = (__bf16*)ws; ws += (size_t)128 * 512 * 2;
  __bf16* Hmu     = (__bf16*)ws; ws += (size_t)512 * 512 * 2;
  __bf16* Hlv     = (__bf16*)ws; ws += (size_t)512 * 512 * 2;
  float*  mu      = (float*)ws;  ws += (size_t)512 * 128 * 4;
  float*  lv      = (float*)ws;  ws += (size_t)512 * 128 * 4;
  __bf16* Abuf    = (__bf16*)ws; ws += (size_t)512 * 256 * 2;
  __bf16* Bbuf    = (__bf16*)ws; ws += (size_t)512 * 256 * 2;
  float*  cvec    = (float*)ws;  ws += (size_t)512 * 4;
  float*  apT     = (float*)ws;  ws += (size_t)512 * 512 * 4;
  float*  contrib = (float*)ws;  ws += (size_t)512 * 4;

  // one-shot operand conversion (weights reused by 32 M-tiles each)
  cvt_bf16<<<(512 * 768 + 255) / 256, 256, 0, stream>>>(x, Xb, 512 * 768);
  transpose_cvt_bf16<<<(768 * 512 + 255) / 256, 256, 0, stream>>>(w1m, W1mt, 768, 512);
  transpose_cvt_bf16<<<(768 * 512 + 255) / 256, 256, 0, stream>>>(w1l, W1lt, 768, 512);
  transpose_cvt_bf16<<<(512 * 128 + 255) / 256, 256, 0, stream>>>(w2m, W2mt, 512, 128);
  transpose_cvt_bf16<<<(512 * 128 + 255) / 256, 256, 0, stream>>>(w2l, W2lt, 512, 128);

  // Layer 1: H = relu(x @ W1 + b1) -> bf16; tiles = 32*32 = 1024, 8 waves/block
  gemm_bf16<0, true><<<128, 256, 0, stream>>>(Xb, W1mt, b1m, nullptr, Hmu, 512, 512, 768);
  gemm_bf16<0, true><<<128, 256, 0, stream>>>(Xb, W1lt, b1l, nullptr, Hlv, 512, 512, 768);
  // Layer 2: mu = H @ W2 + b2 ; logvar = tanh(H @ W2 + b2); tiles = 32*8 = 256
  gemm_bf16<1, false><<<32, 256, 0, stream>>>(Hmu, W2mt, b2m, mu, nullptr, 512, 128, 512);
  gemm_bf16<2, false><<<32, 256, 0, stream>>>(Hlv, W2lt, b2l, lv, nullptr, 512, 128, 512);

  // Contrastive panels + row constants
  prep_contrast<<<512, 128, 0, stream>>>(mu, lv, y, Abuf, Bbuf, cvec);
  // all_probs via single 512x512 (K=256) WMMA GEMM; tiles = 1024
  gemm_ap<<<128, 256, 0, stream>>>(Abuf, Bbuf, cvec, apT);
  // logsumexp over i per column j, then mean
  col_lse<<<512, 256, 0, stream>>>(apT, contrib);
  final_mean<<<1, 256, 0, stream>>>(contrib, (float*)d_out);
}
